// ParaDecoder_48241072668871
// MI455X (gfx1250) — matrix operand order repaired and checked by reading, not compiled, not run on hardware
//
#include <hip/hip_runtime.h>
#include <hip/hip_bf16.h>
#include <math.h>

#define Hd 512
#define Bd 64
#define Td 256
#define Ed 256
#define Vd 4000
#define VXd 4256

typedef __attribute__((ext_vector_type(16))) __bf16 v16bf;
typedef __attribute__((ext_vector_type(16))) unsigned short v16us;
typedef __attribute__((ext_vector_type(8))) float v8f;

static __device__ __forceinline__ unsigned short f2bf(float f) {
  unsigned u = __float_as_uint(f);
  unsigned r = u + 0x7FFFu + ((u >> 16) & 1u);   // round-to-nearest-even
  return (unsigned short)(r >> 16);
}

// ---------------------------------------------------------------------------
// Convert a 512x512 fp32 weight block (possibly a column slice of a wider
// matrix) to bf16 bits, row-major [j][k].
// ---------------------------------------------------------------------------
__global__ void convert_w_kernel(const float* __restrict__ W,
                                 unsigned short* __restrict__ out,
                                 int row_stride, int col_off) {
  int idx = blockIdx.x * 256 + threadIdx.x;
  if (idx >= Hd * Hd) return;
  int j = idx >> 9, k = idx & (Hd - 1);
  out[idx] = f2bf(W[(size_t)j * row_stride + col_off + k]);
}

// ---------------------------------------------------------------------------
// Fused energy+score epilogue helper: part[i] += tanh(c[i]+bias)*weight
// ---------------------------------------------------------------------------
__device__ __forceinline__ void score_epilogue(
    const v8f& c, int jg, int b0, int lh,
    const float* __restrict__ bias_mat, int bias_perb,
    const float* __restrict__ wmat, int w_perb, float part[8]) {
#pragma unroll
  for (int i = 0; i < 8; ++i) {
    int bi = b0 + 8 * lh + i;
    float bs = bias_perb ? bias_mat[(size_t)bi * Hd + jg] : bias_mat[jg];
    float ww = w_perb ? wmat[(size_t)bi * Hd + jg] : wmat[jg];
    part[i] += tanhf(c[i] + bs) * ww;
  }
}

// ---------------------------------------------------------------------------
// Fused: scores[b][t] = sum_col tanh((enc @ Wb^T)[r][col] + bias) * weight
// enc: (T*B, H) fp32 (row r = t*B+b), Wb: (H,H) bf16 bits.
// One wave handles a 16-row strip across all 512 output columns (K=512).
// Two column tiles are processed per pass with the B fragments for k-step
// kk+1 prefetched before the WMMAs consuming k-step kk (overlaps loads
// with matrix math instead of load->wait0->wmma serialization).
// ---------------------------------------------------------------------------
__global__ __launch_bounds__(256) void energy_score_kernel(
    const float* __restrict__ enc, const unsigned short* __restrict__ Wb,
    const float* __restrict__ bias_mat, int bias_perb,
    const float* __restrict__ wmat, int w_perb,
    float* __restrict__ scores) {
  const int lane = threadIdx.x & 31;
  const int wvid = threadIdx.x >> 5;
  const int strip = blockIdx.x * 8 + wvid;   // 1024 strips total
  const int r0 = strip << 4;
  const int m = lane & 15;
  const int lh = lane >> 4;                  // lane half
  const int b0 = r0 & (Bd - 1);              // strip stays within one t block
  const int t0 = r0 >> 6;

  // A fragments: 16 rows x K=512, bf16 WMMA layout, converted in-register.
  v16bf a[16];
  const float* arow = enc + (size_t)(r0 + m) * Hd + lh * 16;
#pragma unroll
  for (int kk = 0; kk < 16; ++kk) {
    const float4* p = (const float4*)(arow + kk * 32);
    const float4* q = (const float4*)(arow + kk * 32 + 8);
    float4 x0 = p[0], x1 = p[1], y0 = q[0], y1 = q[1];
    v16us u;
    u[0] = f2bf(x0.x);  u[1] = f2bf(x0.y);  u[2] = f2bf(x0.z);  u[3] = f2bf(x0.w);
    u[4] = f2bf(x1.x);  u[5] = f2bf(x1.y);  u[6] = f2bf(x1.z);  u[7] = f2bf(x1.w);
    u[8] = f2bf(y0.x);  u[9] = f2bf(y0.y);  u[10] = f2bf(y0.z); u[11] = f2bf(y0.w);
    u[12] = f2bf(y1.x); u[13] = f2bf(y1.y); u[14] = f2bf(y1.z); u[15] = f2bf(y1.w);
    a[kk] = __builtin_bit_cast(v16bf, u);
  }

  const int col = lane & 15;
  float part[8];
#pragma unroll
  for (int i = 0; i < 8; ++i) part[i] = 0.f;

  for (int jt = 0; jt < 32; jt += 2) {       // 2 column tiles of 16 per pass
    const int jg0 = jt * 16 + col;
    const unsigned short* w0 = Wb + (size_t)jg0 * Hd + lh * 16;
    const unsigned short* w1 = w0 + 16 * Hd;
    v8f c0 = {}, c1 = {};
    v16us b0v = *(const v16us*)(w0);
    v16us b1v = *(const v16us*)(w1);
#pragma unroll
    for (int kk = 0; kk < 16; ++kk) {
      v16us n0 = b0v, n1 = b1v;
      if (kk < 15) {                         // prefetch next k-step fragments
        n0 = *(const v16us*)(w0 + (kk + 1) * 32);
        n1 = *(const v16us*)(w1 + (kk + 1) * 32);
      }
      c0 = __builtin_amdgcn_wmma_f32_16x16x32_bf16(
               false, a[kk], false, __builtin_bit_cast(v16bf, b0v),
               (short)0, c0, false, false);
      c1 = __builtin_amdgcn_wmma_f32_16x16x32_bf16(
               false, a[kk], false, __builtin_bit_cast(v16bf, b1v),
               (short)0, c1, false, false);
      b0v = n0; b1v = n1;
    }
    score_epilogue(c0, jg0, b0, lh, bias_mat, bias_perb, wmat, w_perb, part);
    score_epilogue(c1, jg0 + 16, b0, lh, bias_mat, bias_perb, wmat, w_perb, part);
  }
  // Reduce across the 16 column-lanes of each lane-half.
#pragma unroll
  for (int off = 1; off < 16; off <<= 1) {
#pragma unroll
    for (int i = 0; i < 8; ++i) part[i] += __shfl_xor(part[i], off, 32);
  }
  if (col == 0) {
#pragma unroll
    for (int i = 0; i < 8; ++i)
      scores[(size_t)(b0 + 8 * lh + i) * Td + t0] = part[i];
  }
}

// ---------------------------------------------------------------------------
// Per-batch softmax over T, then ctx[b][h] = sum_t alpha[t] * enc[t][b][h].
// ---------------------------------------------------------------------------
__global__ __launch_bounds__(256) void attn_ctx_kernel(
    const float* __restrict__ scores, const float* __restrict__ enc,
    float* __restrict__ ctx) {
  int b = blockIdx.x, tid = threadIdx.x;
  __shared__ float al[Td];
  __shared__ float red[256];
  float s = scores[b * Td + tid];
  red[tid] = s; __syncthreads();
  for (int off = 128; off > 0; off >>= 1) {
    if (tid < off) red[tid] = fmaxf(red[tid], red[tid + off]);
    __syncthreads();
  }
  float mx = red[0]; __syncthreads();
  float e = __expf(s - mx);
  al[tid] = e; red[tid] = e; __syncthreads();
  for (int off = 128; off > 0; off >>= 1) {
    if (tid < off) red[tid] += red[tid + off];
    __syncthreads();
  }
  float inv = 1.f / red[0];
  float a0 = 0.f, a1 = 0.f;
  for (int t = 0; t < Td; ++t) {
    float av = al[t];
    const float* er = enc + ((size_t)t * Bd + b) * Hd;
    a0 += av * er[tid];
    a1 += av * er[tid + 256];
  }
  ctx[(size_t)b * Hd + tid] = a0 * inv;
  ctx[(size_t)b * Hd + tid + 256] = a1 * inv;
}

// ---------------------------------------------------------------------------
// gru_in[b] = [emb[p[b]] (256) | ctx_u[b] (512) | ctx_a[b] (512)]
// ---------------------------------------------------------------------------
__global__ void gruin_kernel(const int* __restrict__ p, const float* __restrict__ emb,
                             const float* __restrict__ cu, const float* __restrict__ ca,
                             float* __restrict__ gin) {
  int idx = blockIdx.x * 256 + threadIdx.x;
  if (idx >= Bd * 1280) return;
  int b = idx / 1280, k = idx % 1280;
  float v;
  if (k < Ed)           v = emb[(size_t)p[b] * Ed + k];
  else if (k < Ed + Hd) v = cu[b * Hd + (k - Ed)];
  else                  v = ca[b * Hd + (k - Ed - Hd)];
  gin[idx] = v;
}

// ---------------------------------------------------------------------------
// out[M,N] = X[M,K] @ W[N,K(ldw)]^T + bias   (K multiple of 32), LDS-tiled.
// ldw = row stride of W (>= K), so column slices of wider matrices work.
// ---------------------------------------------------------------------------
__global__ __launch_bounds__(256) void sgemm_nt_kernel(
    const float* __restrict__ X, const float* __restrict__ W,
    const float* __restrict__ bias, float* __restrict__ out,
    int M, int N, int K, int ldw, int col_off) {
  __shared__ float Xs[64][33];
  __shared__ float Ws[64][33];
  int tid = threadIdx.x;
  int n0 = blockIdx.x * 64, m0 = blockIdx.y * 64;
  int tx = tid & 15, ty = tid >> 4;
  float acc[4][4] = {};
  for (int k0 = 0; k0 < K; k0 += 32) {
    for (int i = tid; i < 64 * 32; i += 256) {
      int r = i >> 5, c = i & 31;
      Xs[r][c] = (m0 + r < M) ? X[(size_t)(m0 + r) * K + k0 + c] : 0.f;
      Ws[r][c] = (n0 + r < N) ? W[(size_t)(n0 + r) * ldw + col_off + k0 + c] : 0.f;
    }
    __syncthreads();
    for (int kk = 0; kk < 32; ++kk) {
      float xv[4], wv[4];
#pragma unroll
      for (int i = 0; i < 4; ++i) { xv[i] = Xs[ty * 4 + i][kk]; wv[i] = Ws[tx * 4 + i][kk]; }
#pragma unroll
      for (int i = 0; i < 4; ++i)
#pragma unroll
        for (int j = 0; j < 4; ++j) acc[i][j] += xv[i] * wv[j];
    }
    __syncthreads();
  }
  for (int i = 0; i < 4; ++i) {
    int mr = m0 + ty * 4 + i;
    if (mr >= M) continue;
    for (int j = 0; j < 4; ++j) {
      int nc = n0 + tx * 4 + j;
      if (nc < N) out[(size_t)mr * N + nc] = acc[i][j] + bias[nc];
    }
  }
}

// ---------------------------------------------------------------------------
// GRU gate combine; also writes gru_out and new_hidden sections of d_out.
// ---------------------------------------------------------------------------
__global__ void gru_combine_kernel(const float* __restrict__ gi, const float* __restrict__ gh,
                                   const float* __restrict__ hold,
                                   float* __restrict__ hnew, float* __restrict__ out) {
  int idx = blockIdx.x * 256 + threadIdx.x;
  if (idx >= Bd * Hd) return;
  int b = idx >> 9, j = idx & (Hd - 1);
  const float* gib = gi + (size_t)b * 1536;
  const float* ghb = gh + (size_t)b * 1536;
  float r = 1.f / (1.f + __expf(-(gib[j] + ghb[j])));
  float z = 1.f / (1.f + __expf(-(gib[Hd + j] + ghb[Hd + j])));
  float n = tanhf(gib[2 * Hd + j] + r * ghb[2 * Hd + j]);
  float h = (1.f - z) * n + z * hold[idx];
  hnew[idx] = h;
  out[idx] = h;                 // gru_out
  out[Bd * Hd + idx] = h;       // new_hidden
}

// ---------------------------------------------------------------------------
// catv[b] = [h_new[b] | ctx_u[b] | ctx_a[b]]  (B, 1536)
// ---------------------------------------------------------------------------
__global__ void catv_kernel(const float* __restrict__ hn, const float* __restrict__ cu,
                            const float* __restrict__ ca, float* __restrict__ catv) {
  int idx = blockIdx.x * 256 + threadIdx.x;
  if (idx >= Bd * 1536) return;
  int b = idx / 1536, k = idx % 1536;
  float v = (k < Hd) ? hn[b * Hd + k]
          : (k < 2 * Hd) ? cu[b * Hd + (k - Hd)]
          : ca[b * Hd + (k - 2 * Hd)];
  catv[idx] = v;
}

// ---------------------------------------------------------------------------
// u_max[b] = max_t u_copy[b][t];  u_exp[b][t] = exp(u_copy - max)
// ---------------------------------------------------------------------------
__global__ __launch_bounds__(256) void ucopy_prep_kernel(
    const float* __restrict__ uc, float* __restrict__ uexp, float* __restrict__ umax) {
  int b = blockIdx.x, tid = threadIdx.x;
  __shared__ float red[256];
  float s = uc[b * Td + tid];
  red[tid] = s; __syncthreads();
  for (int off = 128; off > 0; off >>= 1) {
    if (tid < off) red[tid] = fmaxf(red[tid], red[tid + off]);
    __syncthreads();
  }
  float mx = red[0];
  uexp[b * Td + tid] = __expf(s - mx);
  if (tid == 0) umax[b] = mx;
}

// ---------------------------------------------------------------------------
// cs[b][v] = log(sum_t u_exp[b][t] * sparse[b][t][v]) + u_max[b]  (279 MB read)
// Sparse tensor is streaming-only and exceeds L2 -> non-temporal loads keep
// enc/weights resident in L2 for the other kernels.
// ---------------------------------------------------------------------------
__global__ __launch_bounds__(256) void copyscore_kernel(
    const float* __restrict__ uexp, const float* __restrict__ umax,
    const float* __restrict__ sparse, float* __restrict__ cs) {
  int b = blockIdx.y;
  int v = blockIdx.x * 256 + threadIdx.x;
  __shared__ float es[Td];
  es[threadIdx.x] = uexp[b * Td + threadIdx.x];
  __syncthreads();
  if (v >= VXd) return;
  const float* sp = sparse + (size_t)b * Td * VXd + v;
  float s = 0.f;
#pragma unroll 4
  for (int t = 0; t < Td; ++t)
    s += es[t] * __builtin_nontemporal_load(&sp[(size_t)t * VXd]);
  cs[(size_t)b * VXd + v] = __logf(s) + umax[b];
}

// ---------------------------------------------------------------------------
// Softmax over concat([gen (V), cs (VX)]) and assemble proba (B, VX) in d_out.
// ---------------------------------------------------------------------------
__global__ __launch_bounds__(256) void final_kernel(
    const float* __restrict__ gen, const float* __restrict__ cs,
    float* __restrict__ out) {
  int b = blockIdx.x, tid = threadIdx.x;
  __shared__ float red[256];
  float mx = -3.4e38f;
  for (int v = tid; v < Vd + VXd; v += 256) {
    float x = (v < Vd) ? gen[(size_t)b * Vd + v] : cs[(size_t)b * VXd + (v - Vd)];
    mx = fmaxf(mx, x);
  }
  red[tid] = mx; __syncthreads();
  for (int off = 128; off > 0; off >>= 1) {
    if (tid < off) red[tid] = fmaxf(red[tid], red[tid + off]);
    __syncthreads();
  }
  mx = red[0]; __syncthreads();
  float sum = 0.f;
  for (int v = tid; v < Vd + VXd; v += 256) {
    float x = (v < Vd) ? gen[(size_t)b * Vd + v] : cs[(size_t)b * VXd + (v - Vd)];
    sum += __expf(x - mx);
  }
  red[tid] = sum; __syncthreads();
  for (int off = 128; off > 0; off >>= 1) {
    if (tid < off) red[tid] += red[tid + off];
    __syncthreads();
  }
  float inv = 1.f / red[0];
  float* pb = out + 2 * Bd * Hd + (size_t)b * VXd;
  for (int v = tid; v < VXd; v += 256) {
    float cp = __expf(cs[(size_t)b * VXd + v] - mx) * inv;
    float p = (v < Vd) ? (__expf(gen[(size_t)b * Vd + v] - mx) * inv + cp) : cp;
    pb[v] = p;
  }
}

// ---------------------------------------------------------------------------
extern "C" void kernel_launch(void* const* d_in, const int* in_sizes, int n_in,
                              void* d_out, int out_size, void* d_ws, size_t ws_size,
                              hipStream_t stream) {
  (void)in_sizes; (void)n_in; (void)out_size; (void)ws_size;
  const float* a_enc   = (const float*)d_in[0];
  const float* u_enc   = (const float*)d_in[1];
  const int*   p_tm1   = (const int*)d_in[2];
  const float* hidden  = (const float*)d_in[3];
  const float* sparse  = (const float*)d_in[6];
  const float* emb     = (const float*)d_in[7];
  const float* attn_u_W = (const float*)d_in[8];
  const float* attn_u_b = (const float*)d_in[9];
  const float* attn_u_v = (const float*)d_in[10];
  const float* attn_a_W = (const float*)d_in[11];
  const float* attn_a_b = (const float*)d_in[12];
  const float* attn_a_v = (const float*)d_in[13];
  const float* W_ih   = (const float*)d_in[14];
  const float* W_hh   = (const float*)d_in[15];
  const float* b_ih   = (const float*)d_in[16];
  const float* b_hh   = (const float*)d_in[17];
  const float* proj_W = (const float*)d_in[18];
  const float* proj_b = (const float*)d_in[19];
  const float* copy_W = (const float*)d_in[20];
  const float* copy_b = (const float*)d_in[21];
  float* out = (float*)d_out;

  char* ws = (char*)d_ws;
  size_t o = 0;
  auto alloc = [&](size_t bytes) -> void* {
    void* p = ws + o;
    o += (bytes + 255) & ~(size_t)255;
    return p;
  };
  unsigned short* WBU = (unsigned short*)alloc(Hd * Hd * 2);
  unsigned short* WBA = (unsigned short*)alloc(Hd * Hd * 2);
  unsigned short* WBC = (unsigned short*)alloc(Hd * Hd * 2);
  float* HBU  = (float*)alloc(Bd * Hd * 4);
  float* HBA  = (float*)alloc(Bd * Hd * 4);
  float* SCU  = (float*)alloc(Bd * Td * 4);
  float* SCA  = (float*)alloc(Bd * Td * 4);
  float* CTU  = (float*)alloc(Bd * Hd * 4);
  float* CTA  = (float*)alloc(Bd * Hd * 4);
  float* GIN  = (float*)alloc(Bd * 1280 * 4);
  float* GI   = (float*)alloc(Bd * 1536 * 4);
  float* GH   = (float*)alloc(Bd * 1536 * 4);
  float* HNEW = (float*)alloc(Bd * Hd * 4);
  float* CATV = (float*)alloc(Bd * 1536 * 4);
  float* GEN  = (float*)alloc((size_t)Bd * Vd * 4);
  float* UCP  = (float*)alloc(Bd * Td * 4);
  float* UEX  = (float*)alloc(Bd * Td * 4);
  float* UMX  = (float*)alloc(Bd * 4);
  float* CS   = (float*)alloc((size_t)Bd * VXd * 4);

  // Weight bf16 conversion (B-fragment friendly, row-major [j][k])
  convert_w_kernel<<<Hd * Hd / 256, 256, 0, stream>>>(attn_u_W, WBU, 2 * Hd, Hd);
  convert_w_kernel<<<Hd * Hd / 256, 256, 0, stream>>>(attn_a_W, WBA, 2 * Hd, Hd);
  convert_w_kernel<<<Hd * Hd / 256, 256, 0, stream>>>(copy_W,  WBC, Hd, 0);

  // Per-batch hidden-projection biases: hbias = hidden @ W[:, :H].T + b
  sgemm_nt_kernel<<<dim3(Hd / 64, 1), 256, 0, stream>>>(hidden, attn_u_W, attn_u_b, HBU,
                                                        Bd, Hd, Hd, 2 * Hd, 0);
  sgemm_nt_kernel<<<dim3(Hd / 64, 1), 256, 0, stream>>>(hidden, attn_a_W, attn_a_b, HBA,
                                                        Bd, Hd, Hd, 2 * Hd, 0);

  // Fused WMMA energy matmul + tanh + dot(v) -> attention scores
  energy_score_kernel<<<128, 256, 0, stream>>>(u_enc, WBU, HBU, 1, attn_u_v, 0, SCU);
  energy_score_kernel<<<128, 256, 0, stream>>>(a_enc, WBA, HBA, 1, attn_a_v, 0, SCA);

  // Softmax + context
  attn_ctx_kernel<<<Bd, 256, 0, stream>>>(SCU, u_enc, CTU);
  attn_ctx_kernel<<<Bd, 256, 0, stream>>>(SCA, a_enc, CTA);

  // GRU
  gruin_kernel<<<(Bd * 1280 + 255) / 256, 256, 0, stream>>>(p_tm1, emb, CTU, CTA, GIN);
  sgemm_nt_kernel<<<dim3((1536 + 63) / 64, 1), 256, 0, stream>>>(GIN, W_ih, b_ih, GI,
                                                                 Bd, 1536, 1280, 1280, 0);
  sgemm_nt_kernel<<<dim3((1536 + 63) / 64, 1), 256, 0, stream>>>(hidden, W_hh, b_hh, GH,
                                                                 Bd, 1536, Hd, Hd, 0);
  gru_combine_kernel<<<(Bd * Hd + 255) / 256, 256, 0, stream>>>(GI, GH, hidden, HNEW, out);

  // Fused WMMA copy-energy matmul + tanh + dot(h_new[b]) -> u_copy scores
  energy_score_kernel<<<128, 256, 0, stream>>>(u_enc, WBC, copy_b, 0, HNEW, 1, UCP);

  // Projection scores
  catv_kernel<<<(Bd * 1536 + 255) / 256, 256, 0, stream>>>(HNEW, CTU, CTA, CATV);
  sgemm_nt_kernel<<<dim3((Vd + 63) / 64, 1), 256, 0, stream>>>(CATV, proj_W, proj_b, GEN,
                                                               Bd, Vd, 1536, 1536, 0);

  // Copy distribution over sparse vocab map
  ucopy_prep_kernel<<<Bd, 256, 0, stream>>>(UCP, UEX, UMX);
  copyscore_kernel<<<dim3((VXd + 255) / 256, Bd), 256, 0, stream>>>(UEX, UMX, sparse, CS);

  // Final joint softmax + proba assembly
  final_kernel<<<Bd, 256, 0, stream>>>(GEN, CS, out);
}
